// AttributeRoutingModule_30253749633495
// MI455X (gfx1250) — matrix operand order repaired and checked by reading, not compiled, not run on hardware
//
#include <hip/hip_runtime.h>

// ---- problem constants ----
#define B_SZ   8192
#define A_SZ   8
#define C_SZ   12
#define D_IN   1024
#define D_OUT  1024
#define N_FLAT (C_SZ * D_OUT)        // 12288
#define N_ITERS 3
#define ENTROPY_REG 0.4f

typedef __attribute__((ext_vector_type(16))) __bf16 v16bf;
typedef __attribute__((ext_vector_type(8)))  float  v8f;

// =====================================================================
// Kernel 1: f32 -> bf16 conversion (grid-stride)
// =====================================================================
__global__ void cvt_f32_bf16(const float* __restrict__ in,
                             __bf16* __restrict__ out, long n) {
    long i = (long)blockIdx.x * blockDim.x + threadIdx.x;
    long stride = (long)gridDim.x * blockDim.x;
    for (; i < n; i += stride) out[i] = (__bf16)in[i];
}

// =====================================================================
// Kernel 2: GEMM  U[8192][12288] = X[8192][1024] @ Wf[12288][1024]^T + bias
// bf16 in, f32 accumulate (v_wmma_f32_16x16x32_bf16), bf16 out.
// 128x128 workgroup tile, K-step 32, double-buffered LDS filled with
// global_load_async_to_lds_b128 (ASYNCcnt) so global latency overlaps the
// WMMA burst. 8 waves; wave w owns rows [w*16, w*16+16) x 128 cols.
// =====================================================================
#define LDSTR 40   // bf16 elems per LDS row (32 + 8 pad; 80B -> odd bank stride)

// LDS byte offset of a flat shared pointer: aperture model puts the LDS
// offset in addr[31:0]  (CDNA5 ISA 10.2 aperture calc).
__device__ __forceinline__ unsigned lds_off(const void* p) {
    return (unsigned)(unsigned long long)p;
}

__device__ __forceinline__ void async_stage_tile(const __bf16* __restrict__ X,
                                                 const __bf16* __restrict__ Wf,
                                                 __bf16* sA, __bf16* sB,
                                                 int tid, int m0, int n0, int k0) {
    // 128x32 bf16 tile = 512 chunks of 16B; 2 chunks/thread/matrix
    #pragma unroll
    for (int i = 0; i < 2; ++i) {
        int q   = tid + i * 256;
        int row = q >> 2;
        int cp  = q & 3;
        unsigned       la = lds_off(sA + row * LDSTR + cp * 8);
        const __bf16*  ga = X + (size_t)(m0 + row) * D_IN + k0 + cp * 8;
        asm volatile("global_load_async_to_lds_b128 %0, %1, off"
                     :: "v"(la), "v"(ga) : "memory");
        unsigned       lb = lds_off(sB + row * LDSTR + cp * 8);
        const __bf16*  gb = Wf + (size_t)(n0 + row) * D_IN + k0 + cp * 8;
        asm volatile("global_load_async_to_lds_b128 %0, %1, off"
                     :: "v"(lb), "v"(gb) : "memory");
    }
}

__global__ __launch_bounds__(256)
void gemm_bf16_wmma(const __bf16* __restrict__ X,
                    const __bf16* __restrict__ Wf,
                    const float*  __restrict__ bias,
                    __bf16* __restrict__ U) {
    __shared__ __align__(16) __bf16 lA[2][128 * LDSTR];
    __shared__ __align__(16) __bf16 lB[2][128 * LDSTR];

    const int tid  = threadIdx.x;
    const int wave = tid >> 5;
    const int lane = tid & 31;
    const int m0 = blockIdx.y * 128;
    const int n0 = blockIdx.x * 128;

    v8f acc[8] = {};

    const int ml = lane & 15;   // row/col within 16
    const int h  = lane >> 4;   // half-wave select

    union frag { v16bf v; uint4 u[2]; };

    // prologue: async-fill buffer 0 with K-tile 0
    async_stage_tile(X, Wf, lA[0], lB[0], tid, m0, n0, 0);

    for (int kt = 0; kt < D_IN / 32; ++kt) {
        const int cur = kt & 1;

        if (kt + 1 < D_IN / 32) {
            // overlap: start fetching K-tile kt+1 into the other buffer
            async_stage_tile(X, Wf, lA[cur ^ 1], lB[cur ^ 1], tid, m0, n0,
                             (kt + 1) * 32);
            // async loads retire in order: <=4 outstanding => tile kt landed
            asm volatile("s_wait_asynccnt 4" ::: "memory");
        } else {
            asm volatile("s_wait_asynccnt 0" ::: "memory");
        }
        __syncthreads();

        // ---- A fragment (16x32): lanes 0-15: K 0-7 & 16-23; lanes 16-31: +8
        frag af;
        const __bf16* arow = lA[cur] + (wave * 16 + ml) * LDSTR;
        af.u[0] = *(const uint4*)(arow + 8 * h);
        af.u[1] = *(const uint4*)(arow + 16 + 8 * h);

        // ---- preload all 8 B fragments (distinct regs -> one dscnt wait,
        //      then 8 back-to-back v_wmma) ----
        frag bfr[8];
        #pragma unroll
        for (int t = 0; t < 8; ++t) {
            const __bf16* brow = lB[cur] + (t * 16 + ml) * LDSTR;
            bfr[t].u[0] = *(const uint4*)(brow + 8 * h);
            bfr[t].u[1] = *(const uint4*)(brow + 16 + 8 * h);
        }
        #pragma unroll
        for (int t = 0; t < 8; ++t) {
            acc[t] = __builtin_amdgcn_wmma_f32_16x16x32_bf16(
                false, af.v, false, bfr[t].v, (short)0, acc[t], false, false);
        }
        __syncthreads();   // all waves done reading before buffer reuse
    }

    // ---- epilogue: + bias, store bf16 ----
    // C/D layout: VGPR r -> M = r (lanes 0-15) or 8+r (lanes 16-31); N = ml
    #pragma unroll
    for (int t = 0; t < 8; ++t) {
        int col = n0 + t * 16 + ml;
        float bv = bias[col];
        #pragma unroll
        for (int r = 0; r < 8; ++r) {
            int row = m0 + wave * 16 + r + 8 * h;
            U[(size_t)row * N_FLAT + col] = (__bf16)(acc[t][r] + bv);
        }
    }
}

// =====================================================================
// Kernel 3: dynamic routing per sample. One 256-thread WG per b.
// v identical across A (b starts at 0, u_e broadcast) -> compute once,
// replicate 8x. Deterministic reductions (shfl_xor + fixed-order LDS sums).
// =====================================================================
__global__ __launch_bounds__(256)
void routing_kernel(const __bf16* __restrict__ U,
                    float* __restrict__ out,       // [B][A][D_OUT]
                    float* __restrict__ ent) {     // [B]
    const int b    = blockIdx.x;
    const int tid  = threadIdx.x;
    const int lane = tid & 31;
    const int wave = tid >> 5;

    __shared__ float lred[8];
    __shared__ float l12[C_SZ * 8];
    __shared__ float b12[C_SZ];
    __shared__ float bcast;

    // each thread owns 4 columns: col = tid + 256*j
    float u[C_SZ][4];
    #pragma unroll
    for (int c = 0; c < C_SZ; ++c)
        #pragma unroll
        for (int j = 0; j < 4; ++j)
            u[c][j] = (float)U[(size_t)b * N_FLAT + c * D_OUT + tid + 256 * j];

    float blog[C_SZ];
    #pragma unroll
    for (int c = 0; c < C_SZ; ++c) blog[c] = 0.f;

    float s[4], v[4];

    for (int it = 0; it < N_ITERS; ++it) {
        // softmax over C (uniform per thread)
        float bmax = blog[0];
        #pragma unroll
        for (int c = 1; c < C_SZ; ++c) bmax = fmaxf(bmax, blog[c]);
        float cc[C_SZ], csum = 0.f;
        #pragma unroll
        for (int c = 0; c < C_SZ; ++c) { cc[c] = __expf(blog[c] - bmax); csum += cc[c]; }
        float inv = 1.f / csum;
        #pragma unroll
        for (int c = 0; c < C_SZ; ++c) cc[c] *= inv;

        // s = sum_c cc[c] * u[c]
        #pragma unroll
        for (int j = 0; j < 4; ++j) {
            float a = 0.f;
            #pragma unroll
            for (int c = 0; c < C_SZ; ++c) a = fmaf(cc[c], u[c][j], a);
            s[j] = a;
        }

        // ||s||^2 across 1024 cols
        float p = s[0]*s[0] + s[1]*s[1] + s[2]*s[2] + s[3]*s[3];
        for (int o = 16; o; o >>= 1) p += __shfl_xor(p, o, 32);
        if (lane == 0) lred[wave] = p;
        __syncthreads();
        if (tid == 0) { float t = 0.f; for (int w = 0; w < 8; ++w) t += lred[w]; bcast = t; }
        __syncthreads();
        float n2 = bcast;
        __syncthreads();

        float nrm   = sqrtf(n2);
        float scale = n2 / (1.f + n2) / (nrm + 1e-8f);
        #pragma unroll
        for (int j = 0; j < 4; ++j) v[j] = scale * s[j];

        if (it < N_ITERS - 1) {
            // b[c] += dot(u[c], v)
            #pragma unroll
            for (int c = 0; c < C_SZ; ++c) {
                float d = 0.f;
                #pragma unroll
                for (int j = 0; j < 4; ++j) d = fmaf(u[c][j], v[j], d);
                for (int o = 16; o; o >>= 1) d += __shfl_xor(d, o, 32);
                if (lane == 0) l12[c * 8 + wave] = d;
            }
            __syncthreads();
            if (tid < C_SZ) {
                float t = 0.f;
                for (int w = 0; w < 8; ++w) t += l12[tid * 8 + w];
                b12[tid] = t;
            }
            __syncthreads();
            #pragma unroll
            for (int c = 0; c < C_SZ; ++c) blog[c] += b12[c];
            __syncthreads();
        }
    }

    // write v replicated over A heads (identical by construction)
    #pragma unroll
    for (int a = 0; a < A_SZ; ++a)
        #pragma unroll
        for (int j = 0; j < 4; ++j)
            out[((size_t)b * A_SZ + a) * D_OUT + tid + 256 * j] = v[j];

    // entropy of softmax(s) over 1024 cols
    float m = fmaxf(fmaxf(s[0], s[1]), fmaxf(s[2], s[3]));
    for (int o = 16; o; o >>= 1) m = fmaxf(m, __shfl_xor(m, o, 32));
    if (lane == 0) lred[wave] = m;
    __syncthreads();
    if (tid == 0) { float t = lred[0]; for (int w = 1; w < 8; ++w) t = fmaxf(t, lred[w]); bcast = t; }
    __syncthreads();
    float smax = bcast;
    __syncthreads();

    float e[4], esum = 0.f;
    #pragma unroll
    for (int j = 0; j < 4; ++j) { e[j] = __expf(s[j] - smax); esum += e[j]; }
    for (int o = 16; o; o >>= 1) esum += __shfl_xor(esum, o, 32);
    if (lane == 0) lred[wave] = esum;
    __syncthreads();
    if (tid == 0) { float t = 0.f; for (int w = 0; w < 8; ++w) t += lred[w]; bcast = t; }
    __syncthreads();
    float Z = bcast + 1e-10f;
    __syncthreads();

    float hs = 0.f;
    #pragma unroll
    for (int j = 0; j < 4; ++j) { float pj = e[j] / Z; hs -= pj * __logf(pj + 1e-10f); }
    for (int o = 16; o; o >>= 1) hs += __shfl_xor(hs, o, 32);
    if (lane == 0) lred[wave] = hs;
    __syncthreads();
    if (tid == 0) {
        float t = 0.f;
        for (int w = 0; w < 8; ++w) t += lred[w];
        ent[b] = fminf(fmaxf(t, 0.f), 10.f);
    }
}

// =====================================================================
// Kernel 4: loss = -sigmoid(mean(entropy)) * 0.4  (deterministic reduce)
// =====================================================================
__global__ __launch_bounds__(256)
void finalize_loss(const float* __restrict__ ent, float* __restrict__ out_loss) {
    __shared__ float lred[256];
    int tid = threadIdx.x;
    float t = 0.f;
    for (int k = tid; k < B_SZ; k += 256) t += ent[k];
    lred[tid] = t;
    __syncthreads();
    for (int s = 128; s; s >>= 1) {
        if (tid < s) lred[tid] += lred[tid + s];
        __syncthreads();
    }
    if (tid == 0) {
        float mean = lred[0] / (float)B_SZ;   // entropy identical across A
        float sig  = 1.f / (1.f + __expf(-mean));
        *out_loss = -sig * ENTROPY_REG;
    }
}

// =====================================================================
extern "C" void kernel_launch(void* const* d_in, const int* in_sizes, int n_in,
                              void* d_out, int out_size, void* d_ws, size_t ws_size,
                              hipStream_t stream) {
    const float* x    = (const float*)d_in[0];   // [8192][1024]
    const float* W    = (const float*)d_in[1];   // [12][1024][1024] == [12288][1024]
    const float* bias = (const float*)d_in[2];   // [12288]
    float* out = (float*)d_out;                  // [8192][8][1024] + 1 scalar

    char* ws = (char*)d_ws;
    __bf16* xb = (__bf16*)ws;                                    // 16 MiB
    __bf16* wb = (__bf16*)(ws + (size_t)16 * 1024 * 1024);       // 24 MiB
    __bf16* ub = (__bf16*)(ws + (size_t)40 * 1024 * 1024);       // 192 MiB
    float*  ent = (float*)(ws + (size_t)232 * 1024 * 1024);      // 32 KiB

    cvt_f32_bf16<<<2048, 256, 0, stream>>>(x, xb, (long)B_SZ * D_IN);
    cvt_f32_bf16<<<2048, 256, 0, stream>>>(W, wb, (long)N_FLAT * D_IN);

    dim3 ggrid(N_FLAT / 128, B_SZ / 128);   // (96, 64)
    gemm_bf16_wmma<<<ggrid, 256, 0, stream>>>(xb, wb, bias, ub);

    routing_kernel<<<B_SZ, 256, 0, stream>>>(ub, out, ent);

    float* loss_slot = out + (size_t)B_SZ * A_SZ * D_OUT;
    finalize_loss<<<1, 256, 0, stream>>>(ent, loss_slot);
}
